// SwinBlock_17265768529941
// MI455X (gfx1250) — matrix operand order repaired and checked by reading, not compiled, not run on hardware
//
#include <hip/hip_runtime.h>
#include <hip/hip_bf16.h>

// ---------------------------------------------------------------- constants
static constexpr int kB       = 64;
static constexpr int kHW      = 56;
static constexpr int kDIM     = 384;
static constexpr int kHEADS   = 12;
static constexpr int kHDIM    = 32;
static constexpr int kMLP     = 1536;
static constexpr int kNW      = 8;                       // windows per side
static constexpr int kQKV_N   = 3 * kDIM;                // 1152
static constexpr long kTOK    = (long)kB * kHW * kHW;    // 200704
static constexpr float kNEG   = -1e9f;
static constexpr float kQSCALE = 0.17677669529663687f;   // 32^-0.5

typedef __attribute__((ext_vector_type(16))) _Float16 v16h;
typedef __attribute__((ext_vector_type(8)))  _Float16 v8h;
typedef __attribute__((ext_vector_type(8)))  float    v8f;
typedef int v4i_vec __attribute__((vector_size(16)));

// ------------------------------------------------- async global->LDS (CDNA5)
#if __has_builtin(__builtin_amdgcn_global_load_async_to_lds_b128) && \
    __has_builtin(__builtin_amdgcn_s_wait_asynccnt)
#define USE_ASYNC_LDS 1
__device__ __forceinline__ void async_cp_b128(void* lds_dst, const void* gsrc) {
  __builtin_amdgcn_global_load_async_to_lds_b128(
      (__attribute__((address_space(1))) v4i_vec*)gsrc,
      (__attribute__((address_space(3))) v4i_vec*)lds_dst,
      0, 0);
}
__device__ __forceinline__ void async_wait0() {
  __builtin_amdgcn_s_wait_asynccnt(0);
}
#else
#define USE_ASYNC_LDS 0
#endif

// ------------------------------------------------------- fragment loaders
// A-matrix 16x32 f16 (ISA 7.12.2): lane L holds row M=L&15; lanes<16 carry
// K {0..7,16..23}, lanes>=16 carry K {8..15,24..31}. Two 16B chunks per lane.
__device__ __forceinline__ v16h load_a_frag(const _Float16* base, int stride) {
  const int lane = threadIdx.x & 31;
  const int r    = lane & 15;
  const int kb   = (lane & 16) ? 8 : 0;
  const v8h lo = *(const v8h*)(base + r * stride + kb);
  const v8h hi = *(const v8h*)(base + r * stride + kb + 16);
  v16h f;
#pragma unroll
  for (int i = 0; i < 8; ++i) { f[i] = lo[i]; f[i + 8] = hi[i]; }
  return f;
}

// B-matrix 32x16 f16: lane L holds column N=L&15; lanes<16 K {0..7,16..23},
// lanes>=16 K {8..15,24..31}. base must point at [n][k] layout (stride in halves).
__device__ __forceinline__ v16h load_b_frag(const _Float16* base, int stride) {
  const int lane = threadIdx.x & 31;
  const int n    = lane & 15;
  const int kb   = (lane & 16) ? 8 : 0;
  const v8h lo = *(const v8h*)(base + n * stride + kb);
  const v8h hi = *(const v8h*)(base + n * stride + kb + 16);
  v16h f;
#pragma unroll
  for (int i = 0; i < 8; ++i) { f[i] = lo[i]; f[i + 8] = hi[i]; }
  return f;
}

// ---------------------------------------------------------------- helpers
// Convert f32 weight [K,N] into transposed f16 [N,K] so GEMM B staging is a
// straight row copy (no in-kernel transpose).
__global__ void cvt_transpose_f16(const float* __restrict__ in,
                                  _Float16* __restrict__ out, int K, int N) {
  int idx = blockIdx.x * 256 + threadIdx.x;
  if (idx < K * N) {
    const int n = idx / K, k = idx - n * K;
    out[idx] = (_Float16)in[(long)k * N + n];
  }
}

// One wave32 per token; 384 channels = 12 per lane.
__global__ __launch_bounds__(256)
void layernorm_f16(const float* __restrict__ x, const float* __restrict__ g,
                   const float* __restrict__ bta, _Float16* __restrict__ out) {
  const int lane = threadIdx.x & 31;
  const int w    = threadIdx.x >> 5;
  const long tok = (long)blockIdx.x * 8 + w;
  const float* p = x + tok * kDIM;
  float vals[12];
  float s = 0.f, sq = 0.f;
#pragma unroll
  for (int j = 0; j < 12; ++j) {
    float t = p[lane + 32 * j];
    vals[j] = t; s += t; sq += t * t;
  }
#pragma unroll
  for (int m = 16; m >= 1; m >>= 1) {
    s  += __shfl_xor(s, m);
    sq += __shfl_xor(sq, m);
  }
  const float mean = s * (1.f / kDIM);
  const float var  = sq * (1.f / kDIM) - mean * mean;
  const float rstd = rsqrtf(var + 1e-5f);
  _Float16* q = out + tok * kDIM;
#pragma unroll
  for (int j = 0; j < 12; ++j) {
    const int c = lane + 32 * j;
    q[c] = (_Float16)((vals[j] - mean) * rstd * g[c] + bta[c]);
  }
}

// ------------------------------------------------------------ tiled GEMM
// C[M,N] = A[M,K] * B[K,N] with B given pre-transposed Bt[N,K].
// 256 threads = 8 waves; block tile 64x64; each wave: 16x32 (two C tiles).
// Double-buffered LDS; async global->LDS when available.
// EPI: 0 = store f16, 1 = +bias +f32 residual -> f32,
//      2 = +bias, GELU -> f16, 3 = store f16 with q-scale on cols < kDIM
template <int EPI>
__global__ __launch_bounds__(256)
void gemm_wmma(const _Float16* __restrict__ A, const _Float16* __restrict__ Bt,
               const float* __restrict__ bias, const float* __restrict__ res,
               float* __restrict__ outF, _Float16* __restrict__ outH,
               int M, int N, int K) {
  __shared__ __align__(16) _Float16 As[2][64][40];   // [m][k]
  __shared__ __align__(16) _Float16 Bs[2][64][40];   // [n][k]

  const int tid  = threadIdx.x;
  const int lane = tid & 31;
  const int w    = tid >> 5;
  const long row0 = (long)blockIdx.y * 64;
  const int  col0 = blockIdx.x * 64;
  const int  wm = (w & 3) * 16;
  const int  wn = (w >> 2) * 32;

  const int sr = tid >> 2;          // 0..63: tile row (A: m, Bt: n)
  const int sc = (tid & 3) * 8;     // k chunk

  const _Float16* gA = A  + (row0 + sr) * (long)K + sc;
  const _Float16* gB = Bt + ((long)col0 + sr) * (long)K + sc;

  const int nk = K / 32;
  v8f acc0 = {}; v8f acc1 = {};

#if USE_ASYNC_LDS
  async_cp_b128(&As[0][sr][sc], gA);
  async_cp_b128(&Bs[0][sr][sc], gB);
  async_wait0();
  __syncthreads();
  for (int kk = 0; kk < nk; ++kk) {
    if (kk + 1 < nk) {
      const int nb = (kk + 1) & 1;
      async_cp_b128(&As[nb][sr][sc], gA + (kk + 1) * 32);
      async_cp_b128(&Bs[nb][sr][sc], gB + (kk + 1) * 32);
    }
    const int cb = kk & 1;
    const v16h af = load_a_frag(&As[cb][wm][0], 40);
    const v16h b0 = load_b_frag(&Bs[cb][wn][0], 40);
    const v16h b1 = load_b_frag(&Bs[cb][wn + 16][0], 40);
    acc0 = __builtin_amdgcn_wmma_f32_16x16x32_f16(false, af, false, b0,
                                                  (short)0, acc0, false, false);
    acc1 = __builtin_amdgcn_wmma_f32_16x16x32_f16(false, af, false, b1,
                                                  (short)0, acc1, false, false);
    if (kk + 1 < nk) {
      async_wait0();
      __syncthreads();
    }
  }
#else
  // software pipeline: regs hold next k-step while WMMA consumes current LDS
  v8h ra = *(const v8h*)(gA);
  v8h rb = *(const v8h*)(gB);
  for (int kk = 0; kk < nk; ++kk) {
    const int cb = kk & 1;
    *(v8h*)&As[cb][sr][sc] = ra;
    *(v8h*)&Bs[cb][sr][sc] = rb;
    __syncthreads();
    if (kk + 1 < nk) {
      ra = *(const v8h*)(gA + (kk + 1) * 32);
      rb = *(const v8h*)(gB + (kk + 1) * 32);
    }
    const v16h af = load_a_frag(&As[cb][wm][0], 40);
    const v16h b0 = load_b_frag(&Bs[cb][wn][0], 40);
    const v16h b1 = load_b_frag(&Bs[cb][wn + 16][0], 40);
    acc0 = __builtin_amdgcn_wmma_f32_16x16x32_f16(false, af, false, b0,
                                                  (short)0, acc0, false, false);
    acc1 = __builtin_amdgcn_wmma_f32_16x16x32_f16(false, af, false, b1,
                                                  (short)0, acc1, false, false);
  }
#endif

#pragma unroll
  for (int t = 0; t < 2; ++t) {
    const v8f a = t ? acc1 : acc0;
    const int c = col0 + wn + t * 16 + (lane & 15);
#pragma unroll
    for (int v = 0; v < 8; ++v) {
      const long r = row0 + wm + ((lane & 16) ? v + 8 : v);
      float val = a[v];
      if (EPI == 1 || EPI == 2) val += bias[c];
      if (EPI == 1) {
        outF[r * N + c] = val + res[r * N + c];
      } else if (EPI == 2) {
        const float x3 = val * val * val;
        const float gl = 0.5f * val *
            (1.f + tanhf(0.7978845608028654f * (val + 0.044715f * x3)));
        outH[r * N + c] = (_Float16)gl;
      } else if (EPI == 3) {
        const float sv = (c < kDIM) ? kQSCALE : 1.f;  // pre-scale q
        outH[r * N + c] = (_Float16)(val * sv);
      } else {
        outH[r * N + c] = (_Float16)val;
      }
    }
  }
}

// ------------------------------------------------------ windowed attention
// One block per (batch, window, head); 128 threads = 4 waves, each owning a
// 16-row slab. Cyclic shift folded into gather/scatter indexing. q already
// scaled by 32^-0.5 in the qkv GEMM epilogue.
__global__ __launch_bounds__(128)
void swin_attn(const _Float16* __restrict__ qkv,
               const float* __restrict__ pos_emb,
               _Float16* __restrict__ attn_out) {
  __shared__ __align__(16) _Float16 Qs[64][40];   // [i][d]
  __shared__ __align__(16) _Float16 Ks[64][40];   // [j][d]
  __shared__ __align__(16) _Float16 Vt[32][72];   // [d][j] (B layout for P@V)
  __shared__ __align__(16) _Float16 Ps[64][72];   // softmax probs, A layout

  const int tid  = threadIdx.x;
  const int lane = tid & 31;
  const int w    = tid >> 5;
  int bid = blockIdx.x;
  const int h   = bid % kHEADS; bid /= kHEADS;
  const int win = bid % (kNW * kNW); bid /= (kNW * kNW);
  const int b   = bid;
  const int wy = win >> 3, wx = win & 7;

  for (int rr = tid >> 2; rr < 64; rr += 32) {
    const int ch = (tid & 3) * 8;
    if (rr < 49) {
      const int iy = rr / 7, ix = rr % 7;
      const int y = (wy * 7 + iy + 3) % kHW;      // roll(-3) gather
      const int x = (wx * 7 + ix + 3) % kHW;
      const long tok = ((long)b * kHW + y) * kHW + x;
      const _Float16* base = qkv + tok * kQKV_N + h * kHDIM + ch;
      const v8h qv = *(const v8h*)(base);
      const v8h kv = *(const v8h*)(base + kDIM);
      const v8h vv = *(const v8h*)(base + 2 * kDIM);
      *(v8h*)&Qs[rr][ch] = qv;
      *(v8h*)&Ks[rr][ch] = kv;
#pragma unroll
      for (int i = 0; i < 8; ++i) Vt[ch + i][rr] = vv[i];
    } else {
      const v8h z = {};
      *(v8h*)&Qs[rr][ch] = z;
      *(v8h*)&Ks[rr][ch] = z;
#pragma unroll
      for (int i = 0; i < 8; ++i) Vt[ch + i][rr] = (_Float16)0.f;
    }
  }
  __syncthreads();

  // S = Q K^T : one A-frag, 4 key tiles
  const v16h aq = load_a_frag(&Qs[16 * w][0], 40);
  v8f s[4];
#pragma unroll
  for (int ct = 0; ct < 4; ++ct) {
    const v16h bk = load_b_frag(&Ks[16 * ct][0], 40);
    v8f z = {};
    s[ct] = __builtin_amdgcn_wmma_f32_16x16x32_f16(false, aq, false, bk,
                                                   (short)0, z, false, false);
  }

  // relative-position bias + shift masks + padding
#pragma unroll
  for (int ct = 0; ct < 4; ++ct) {
    const int j = 16 * ct + (lane & 15);
#pragma unroll
    for (int v = 0; v < 8; ++v) {
      const int i = 16 * w + ((lane & 16) ? v + 8 : v);
      float val = s[ct][v];
      if (j >= 49 || i >= 49) {
        val = kNEG;
      } else {
        const int iy = i / 7, ix = i % 7, jy = j / 7, jx = j % 7;
        val += pos_emb[(jy - iy + 6) * 13 + (jx - ix + 6)];
        if (wy == kNW - 1 && ((i >= 28) != (j >= 28)))   val += kNEG;
        if (wx == kNW - 1 && ((ix >= 4) != (jx >= 4)))   val += kNEG;
      }
      s[ct][v] = val;
    }
  }

  // row softmax: each row's 64 cols live across 4 tiles x 16 lanes (half-wave)
#pragma unroll
  for (int v = 0; v < 8; ++v) {
    float m = fmaxf(fmaxf(s[0][v], s[1][v]), fmaxf(s[2][v], s[3][v]));
    m = fmaxf(m, __shfl_xor(m, 1));
    m = fmaxf(m, __shfl_xor(m, 2));
    m = fmaxf(m, __shfl_xor(m, 4));
    m = fmaxf(m, __shfl_xor(m, 8));
    float sum = 0.f;
#pragma unroll
    for (int ct = 0; ct < 4; ++ct) {
      const float e = __expf(s[ct][v] - m);
      s[ct][v] = e; sum += e;
    }
    sum += __shfl_xor(sum, 1);
    sum += __shfl_xor(sum, 2);
    sum += __shfl_xor(sum, 4);
    sum += __shfl_xor(sum, 8);
    const float inv = 1.f / sum;
#pragma unroll
    for (int ct = 0; ct < 4; ++ct) s[ct][v] *= inv;
  }

  // spill P to LDS in A layout
#pragma unroll
  for (int ct = 0; ct < 4; ++ct) {
#pragma unroll
    for (int v = 0; v < 8; ++v) {
      Ps[16 * w + ((lane & 16) ? v + 8 : v)][16 * ct + (lane & 15)] =
          (_Float16)s[ct][v];
    }
  }
  __syncthreads();

  // O = P V : K=64 -> 2 k-steps, N=32 -> 2 tiles
  v8f o0 = {}, o1 = {};
#pragma unroll
  for (int kt = 0; kt < 2; ++kt) {
    const v16h ap = load_a_frag(&Ps[16 * w][kt * 32], 72);
    const v16h b0 = load_b_frag(&Vt[0][kt * 32], 72);
    const v16h b1 = load_b_frag(&Vt[16][kt * 32], 72);
    o0 = __builtin_amdgcn_wmma_f32_16x16x32_f16(false, ap, false, b0,
                                                (short)0, o0, false, false);
    o1 = __builtin_amdgcn_wmma_f32_16x16x32_f16(false, ap, false, b1,
                                                (short)0, o1, false, false);
  }

  // scatter through inverse shift (same index map as gather)
#pragma unroll
  for (int t = 0; t < 2; ++t) {
    const v8f o = t ? o1 : o0;
    const int d = t * 16 + (lane & 15);
#pragma unroll
    for (int v = 0; v < 8; ++v) {
      const int i = 16 * w + ((lane & 16) ? v + 8 : v);
      if (i < 49) {
        const int iy = i / 7, ix = i % 7;
        const int y = (wy * 7 + iy + 3) % kHW;
        const int x = (wx * 7 + ix + 3) % kHW;
        const long tok = ((long)b * kHW + y) * kHW + x;
        attn_out[tok * kDIM + h * kHDIM + d] = (_Float16)o[v];
      }
    }
  }
}

// ---------------------------------------------------------------- launcher
extern "C" void kernel_launch(void* const* d_in, const int* in_sizes, int n_in,
                              void* d_out, int out_size, void* d_ws, size_t ws_size,
                              hipStream_t stream) {
  const float* x      = (const float*)d_in[0];
  const float* w_qkv  = (const float*)d_in[1];
  const float* pos    = (const float*)d_in[2];
  const float* w_out  = (const float*)d_in[3];
  const float* b_out  = (const float*)d_in[4];
  const float* ln1_g  = (const float*)d_in[5];
  const float* ln1_b  = (const float*)d_in[6];
  const float* ln2_g  = (const float*)d_in[7];
  const float* ln2_b  = (const float*)d_in[8];
  const float* w_mlp1 = (const float*)d_in[9];
  const float* b_mlp1 = (const float*)d_in[10];
  const float* w_mlp2 = (const float*)d_in[11];
  const float* b_mlp2 = (const float*)d_in[12];
  float* out = (float*)d_out;

  char* ws = (char*)d_ws;
  auto alloc = [&](size_t bytes) -> char* {
    char* p = ws;
    ws += (bytes + 255) & ~(size_t)255;
    return p;
  };
  _Float16* wqkv_t = (_Float16*)alloc((size_t)kDIM * kQKV_N * 2);  // [1152][384]
  _Float16* wout_t = (_Float16*)alloc((size_t)kDIM * kDIM * 2);    // [384][384]
  _Float16* wm1_t  = (_Float16*)alloc((size_t)kDIM * kMLP * 2);    // [1536][384]
  _Float16* wm2_t  = (_Float16*)alloc((size_t)kMLP * kDIM * 2);    // [384][1536]
  _Float16* bufA   = (_Float16*)alloc((size_t)kTOK * kDIM * 2);    // ln1 / ln2
  char* big        = alloc((size_t)kTOK * kMLP * 2);               // aliased pool
  _Float16* qkv_h  = (_Float16*)big;                                // T x 1152
  _Float16* attn_h = (_Float16*)(big + (size_t)kTOK * kQKV_N * 2); // T x 384
  _Float16* h1_h   = (_Float16*)big;                                // T x 1536

  const int nqkv = kDIM * kQKV_N, nout = kDIM * kDIM,
            nm1 = kDIM * kMLP, nm2 = kMLP * kDIM;
  cvt_transpose_f16<<<(nqkv + 255) / 256, 256, 0, stream>>>(w_qkv, wqkv_t, kDIM, kQKV_N);
  cvt_transpose_f16<<<(nout + 255) / 256, 256, 0, stream>>>(w_out, wout_t, kDIM, kDIM);
  cvt_transpose_f16<<<(nm1 + 255) / 256, 256, 0, stream>>>(w_mlp1, wm1_t, kDIM, kMLP);
  cvt_transpose_f16<<<(nm2 + 255) / 256, 256, 0, stream>>>(w_mlp2, wm2_t, kMLP, kDIM);

  const int mblocks = (int)(kTOK / 64);   // 3136

  // LN1 -> qkv GEMM (q columns pre-scaled by 32^-0.5)
  layernorm_f16<<<(int)(kTOK / 8), 256, 0, stream>>>(x, ln1_g, ln1_b, bufA);
  gemm_wmma<3><<<dim3(kQKV_N / 64, mblocks), 256, 0, stream>>>(
      bufA, wqkv_t, nullptr, nullptr, nullptr, qkv_h, (int)kTOK, kQKV_N, kDIM);

  // shifted-window attention
  swin_attn<<<kB * kNW * kNW * kHEADS, 128, 0, stream>>>(qkv_h, pos, attn_h);

  // output projection + residual with original x
  gemm_wmma<1><<<dim3(kDIM / 64, mblocks), 256, 0, stream>>>(
      attn_h, wout_t, b_out, x, out, nullptr, (int)kTOK, kDIM, kDIM);

  // LN2 -> MLP1 (GELU) -> MLP2 + residual
  layernorm_f16<<<(int)(kTOK / 8), 256, 0, stream>>>(out, ln2_g, ln2_b, bufA);
  gemm_wmma<2><<<dim3(kMLP / 64, mblocks), 256, 0, stream>>>(
      bufA, wm1_t, b_mlp1, nullptr, nullptr, h1_h, (int)kTOK, kMLP, kDIM);
  gemm_wmma<1><<<dim3(kDIM / 64, mblocks), 256, 0, stream>>>(
      h1_h, wm2_t, b_mlp2, out, out, nullptr, (int)kTOK, kDIM, kMLP);
}
